// VITEncoder_78907139162306
// MI455X (gfx1250) — matrix operand order repaired and checked by reading, not compiled, not run on hardware
//
#include <hip/hip_runtime.h>
#include <hip/hip_bf16.h>

// ---------------------------------------------------------------------------
// ViT encoder block (B=16, N=577, D=768, H=12, Dh=64, M=3072) for gfx1250.
// All GEMMs + attention run on v_wmma_f32_16x16x32_bf16 (wave32 WMMA).
// GEMM wave tile: 32 rows x 64 cols (8 WMMAs per 32-k chunk, 12 b128 loads).
// ---------------------------------------------------------------------------

typedef __bf16 bf16_t;
typedef __bf16 v16bf __attribute__((ext_vector_type(16)));
typedef __bf16 v8bf  __attribute__((ext_vector_type(8)));
typedef float  v8f   __attribute__((ext_vector_type(8)));

#define NTOK 577
#define DMODEL 768
#define NHEAD 12
#define DHEAD 64
#define DMLP 3072
#define TTOK 9232          /* 16*577 */
#define TTOKP 9248         /* padded to multiple of 32 for A-side loads */
#define NPAD 608           /* 19*32 key padding */
#define NPADQ 640          /* 10*64 query padding */

// ---- WMMA wrapper: D = A(16x32 bf16) * B(32x16 bf16) + C(f32) -------------
static __device__ inline v8f wmma_bf16(v16bf a, v16bf b, v8f c) {
  return __builtin_amdgcn_wmma_f32_16x16x32_bf16(
      /*neg_a=*/false, a, /*neg_b=*/false, b,
      /*c_mod=*/(short)0, c, /*reuse_a=*/false, /*reuse_b=*/false);
}

// A-operand (16x32, row-major source, ld in elements):
// lane<16: row=lane, K {0..7,16..23}; lane>=16: row=lane-16, K {8..15,24..31}
static __device__ inline v16bf load_a16x32(const bf16_t* __restrict__ A,
                                           int ld, int row0, int k0, int lane) {
  int m   = lane & 15;
  int sel = (lane >> 4) & 1;
  const bf16_t* p = A + (size_t)(row0 + m) * ld + k0 + sel * 8;
  v8bf lo = *(const v8bf*)(p);
  v8bf hi = *(const v8bf*)(p + 16);
  v16bf r;
#pragma unroll
  for (int i = 0; i < 8; ++i) { r[i] = lo[i]; r[i + 8] = hi[i]; }
  return r;
}

// B-operand (32x16). Source is "column-major-by-row": row n0+lane%16 holds the
// contiguous K values of output column n. lane<16: K {0..7,16..23}, lane>=16: K {8..15,24..31}.
static __device__ inline v16bf load_b32x16(const bf16_t* __restrict__ Wt,
                                           int ldk, int n0, int k0, int lane) {
  int n   = lane & 15;
  int sel = (lane >> 4) & 1;
  const bf16_t* p = Wt + (size_t)(n0 + n) * ldk + k0 + sel * 8;
  v8bf lo = *(const v8bf*)(p);
  v8bf hi = *(const v8bf*)(p + 16);
  v16bf r;
#pragma unroll
  for (int i = 0; i < 8; ++i) { r[i] = lo[i]; r[i + 8] = hi[i]; }
  return r;
}

// ---- weight convert+transpose: wt[n*K+k] = (bf16) w[k*N+n] ----------------
__global__ void cvt_t_kernel(const float* __restrict__ w, bf16_t* __restrict__ wt,
                             int K, int N) {
  size_t idx = (size_t)blockIdx.x * 256 + threadIdx.x;
  size_t total = (size_t)K * N;
  if (idx >= total) return;
  size_t n = idx / K, k = idx % K;
  wt[idx] = (bf16_t)w[k * (size_t)N + n];
}

// ---- LayerNorm (one row of D=768 per block) -> bf16 -----------------------
__global__ __launch_bounds__(256)
void ln_kernel(const float* __restrict__ x, const float* __restrict__ g,
               const float* __restrict__ b, bf16_t* __restrict__ out) {
  __shared__ float red[2][8];
  int row = blockIdx.x;
  const float* xr = x + (size_t)row * DMODEL;
  float s = 0.f, s2 = 0.f;
  for (int c = threadIdx.x; c < DMODEL; c += 256) {
    float v = xr[c]; s += v; s2 += v * v;
  }
#pragma unroll
  for (int off = 16; off; off >>= 1) {
    s  += __shfl_xor(s,  off, 32);
    s2 += __shfl_xor(s2, off, 32);
  }
  int wid = threadIdx.x >> 5, lane = threadIdx.x & 31;
  if (lane == 0) { red[0][wid] = s; red[1][wid] = s2; }
  __syncthreads();
  if (threadIdx.x < 32) {
    s  = (lane < 8) ? red[0][lane] : 0.f;
    s2 = (lane < 8) ? red[1][lane] : 0.f;
#pragma unroll
    for (int off = 4; off; off >>= 1) {
      s  += __shfl_xor(s,  off, 32);
      s2 += __shfl_xor(s2, off, 32);
    }
    if (lane == 0) { red[0][0] = s; red[1][0] = s2; }
  }
  __syncthreads();
  float mean = red[0][0] * (1.0f / DMODEL);
  float var  = red[1][0] * (1.0f / DMODEL) - mean * mean;
  float rstd = rsqrtf(var + 1e-5f);
  for (int c = threadIdx.x; c < DMODEL; c += 256) {
    float v = (xr[c] - mean) * rstd * g[c] + b[c];
    out[(size_t)row * DMODEL + c] = (bf16_t)v;
  }
}

// ---- generic bf16 GEMM, wave tile 32x64, 8 waves/block (256 rows x 64 cols)
enum { EPI_QKV = 0, EPI_RESID = 1, EPI_GELU = 2 };

template <int EPI>
__global__ __launch_bounds__(256)
void gemm_kernel(const bf16_t* __restrict__ A, const bf16_t* __restrict__ Wt,
                 const float* __restrict__ bias, int M, int Ncols, int K,
                 const float* __restrict__ resid, float* __restrict__ outF,
                 bf16_t* __restrict__ outH,
                 bf16_t* __restrict__ Qb, bf16_t* __restrict__ Kb,
                 bf16_t* __restrict__ Vt) {
  int lane = threadIdx.x & 31;
  int wid  = threadIdx.x >> 5;
  int row0 = (blockIdx.x * 8 + wid) * 32;
  if (row0 >= M) return;              // wave-uniform; no barriers in kernel
  int nb = blockIdx.y * 64;

  v8f acc[2][4] = {};                 // [row-half][col-tile]
  for (int k0 = 0; k0 < K; k0 += 32) {
    v16bf a0 = load_a16x32(A, K, row0,      k0, lane);
    v16bf a1 = load_a16x32(A, K, row0 + 16, k0, lane);
#pragma unroll
    for (int t = 0; t < 4; ++t) {
      v16bf bm = load_b32x16(Wt, K, nb + t * 16, k0, lane);
      acc[0][t] = wmma_bf16(a0, bm, acc[0][t]);
      acc[1][t] = wmma_bf16(a1, bm, acc[1][t]);
    }
  }

  int colw = lane & 15;
#pragma unroll
  for (int h2 = 0; h2 < 2; ++h2) {
    if (row0 + h2 * 16 >= M) continue;   // wave-uniform guard (M % 16 == 0)
    int rbase = row0 + h2 * 16 + ((lane >> 4) ? 8 : 0);
#pragma unroll
    for (int t = 0; t < 4; ++t) {
      int c = nb + t * 16 + colw;
      float bv = bias[c];
#pragma unroll
      for (int r = 0; r < 8; ++r) {
        int row = rbase + r;
        float v = acc[h2][t][r] + bv;
        if (EPI == EPI_RESID) {
          outF[(size_t)row * Ncols + c] = v + resid[(size_t)row * Ncols + c];
        } else if (EPI == EPI_GELU) {
          float gel = 0.5f * v * (1.0f + erff(v * 0.70710678118654752f));
          outH[(size_t)row * Ncols + c] = (bf16_t)gel;
        } else { // EPI_QKV scatter: Q [BH,640,64], K [BH,608,64], Vt [BH,64,608]
          int bidx = row / NTOK, n = row % NTOK;
          if (c < DMODEL) {
            int h = c >> 6, dh = c & 63;
            Qb[(((size_t)(bidx * NHEAD + h)) * NPADQ + n) * DHEAD + dh] = (bf16_t)v;
          } else if (c < 2 * DMODEL) {
            int c2 = c - DMODEL; int h = c2 >> 6, dh = c2 & 63;
            Kb[(((size_t)(bidx * NHEAD + h)) * NPAD + n) * DHEAD + dh] = (bf16_t)v;
          } else {
            int c2 = c - 2 * DMODEL; int h = c2 >> 6, dh = c2 & 63;
            Vt[(((size_t)(bidx * NHEAD + h)) * DHEAD + dh) * NPAD + n] = (bf16_t)v;
          }
        }
      }
    }
  }
}

// ---- fused flash attention: 1 wave = 16 query rows, Bc = 32 keys/step -----
__global__ __launch_bounds__(128)
void attn_kernel(const bf16_t* __restrict__ Qb, const bf16_t* __restrict__ Kb,
                 const bf16_t* __restrict__ Vt, bf16_t* __restrict__ wa) {
  __shared__ __align__(16) bf16_t plds[4][16 * 32];
  int lane = threadIdx.x & 31;
  int wid  = threadIdx.x >> 5;
  int bh   = blockIdx.x;                    // 0..191
  int b    = bh / NHEAD, h = bh % NHEAD;
  int n0   = blockIdx.y * 64 + wid * 16;    // query tile base (<= 624)

  const bf16_t* Qp = Qb + (size_t)bh * NPADQ * DHEAD;
  const bf16_t* Kp = Kb + (size_t)bh * NPAD * DHEAD;
  const bf16_t* Vp = Vt + (size_t)bh * DHEAD * NPAD;

  v16bf aq0 = load_a16x32(Qp, DHEAD, n0, 0,  lane);
  v16bf aq1 = load_a16x32(Qp, DHEAD, n0, 32, lane);

  v8f o[4] = {};
  float mrow[8], lrow[8];
#pragma unroll
  for (int r = 0; r < 8; ++r) { mrow[r] = -1e30f; lrow[r] = 0.f; }

  const float scale = 0.125f;               // Dh^-0.5
  bf16_t* pl = plds[wid];
  const v8f z = {0.f, 0.f, 0.f, 0.f, 0.f, 0.f, 0.f, 0.f};

  for (int jt = 0; jt < NPAD / 32; ++jt) {
    int jb = jt * 32;
    // ---- S = Q K^T (two 16-key tiles, dh split 0..31 / 32..63) ----
    v16bf bk = load_b32x16(Kp, DHEAD, jb, 0, lane);
    v8f s0 = wmma_bf16(aq0, bk, z);
    bk = load_b32x16(Kp, DHEAD, jb, 32, lane);
    s0 = wmma_bf16(aq1, bk, s0);
    bk = load_b32x16(Kp, DHEAD, jb + 16, 0, lane);
    v8f s1 = wmma_bf16(aq0, bk, z);
    bk = load_b32x16(Kp, DHEAD, jb + 16, 32, lane);
    s1 = wmma_bf16(aq1, bk, s1);

    int c0 = jb + (lane & 15);
    int c1 = c0 + 16;
    float p0[8], p1[8];
#pragma unroll
    for (int r = 0; r < 8; ++r) {
      float v0 = s0[r] * scale; if (c0 >= NTOK) v0 = -1e30f;
      float v1 = s1[r] * scale; if (c1 >= NTOK) v1 = -1e30f;
      // row max over 32 columns (16 lanes of this half-wave hold one row)
      float t = fmaxf(v0, v1);
      t = fmaxf(t, __shfl_xor(t, 1, 32));
      t = fmaxf(t, __shfl_xor(t, 2, 32));
      t = fmaxf(t, __shfl_xor(t, 4, 32));
      t = fmaxf(t, __shfl_xor(t, 8, 32));
      float mn = fmaxf(mrow[r], t);
      float alpha = __expf(mrow[r] - mn);
      mrow[r] = mn;
      p0[r] = __expf(v0 - mn);
      p1[r] = __expf(v1 - mn);
      float rs = p0[r] + p1[r];
      rs += __shfl_xor(rs, 1, 32);
      rs += __shfl_xor(rs, 2, 32);
      rs += __shfl_xor(rs, 4, 32);
      rs += __shfl_xor(rs, 8, 32);
      lrow[r] = lrow[r] * alpha + rs;
#pragma unroll
      for (int t4 = 0; t4 < 4; ++t4) o[t4][r] *= alpha;
    }

    // ---- transpose P (C-layout -> A-layout) through per-wave LDS ----
    {
      int cw = lane & 15;
      int rofs = (lane >> 4) ? 8 : 0;
#pragma unroll
      for (int r = 0; r < 8; ++r) {
        pl[(rofs + r) * 32 + cw]      = (bf16_t)p0[r];
        pl[(rofs + r) * 32 + cw + 16] = (bf16_t)p1[r];
      }
    }
    __builtin_amdgcn_wave_barrier();
    asm volatile("s_wait_dscnt 0" ::: "memory");   // same-wave LDS is in-order
    {
      int m = lane & 15;
      int sel = (lane >> 4) & 1;
      const bf16_t* pp = pl + m * 32 + sel * 8;
      v8bf lo = *(const v8bf*)(pp);
      v8bf hi = *(const v8bf*)(pp + 16);
      v16bf pa;
#pragma unroll
      for (int i = 0; i < 8; ++i) { pa[i] = lo[i]; pa[i + 8] = hi[i]; }
      // ---- O += P V : B-operand from V^T rows (contiguous keys) ----
#pragma unroll
      for (int t4 = 0; t4 < 4; ++t4) {
        v16bf bv = load_b32x16(Vp, NPAD, t4 * 16, jb, lane);
        o[t4] = wmma_bf16(pa, bv, o[t4]);
      }
    }
    __builtin_amdgcn_wave_barrier();
    asm volatile("" ::: "memory");
  }

  // ---- normalize + store (bf16 row-major [TTOK, DMODEL]) ----
  int cw = lane & 15;
  int rofs = (lane >> 4) ? 8 : 0;
#pragma unroll
  for (int t4 = 0; t4 < 4; ++t4) {
    int dh = t4 * 16 + cw;
#pragma unroll
    for (int r = 0; r < 8; ++r) {
      int n = n0 + rofs + r;
      if (n < NTOK) {
        float v = o[t4][r] / lrow[r];
        wa[((size_t)(b * NTOK + n)) * DMODEL + h * DHEAD + dh] = (bf16_t)v;
      }
    }
  }
}

// ---------------------------------------------------------------------------
extern "C" void kernel_launch(void* const* d_in, const int* in_sizes, int n_in,
                              void* d_out, int out_size, void* d_ws, size_t ws_size,
                              hipStream_t stream) {
  (void)in_sizes; (void)n_in; (void)out_size; (void)ws_size;
  const float* x     = (const float*)d_in[0];
  const float* ln1_g = (const float*)d_in[1];
  const float* ln1_b = (const float*)d_in[2];
  const float* w_qkv = (const float*)d_in[3];
  const float* b_qkv = (const float*)d_in[4];
  const float* w_out = (const float*)d_in[5];
  const float* b_out = (const float*)d_in[6];
  const float* ln2_g = (const float*)d_in[7];
  const float* ln2_b = (const float*)d_in[8];
  const float* w_fc1 = (const float*)d_in[9];
  const float* b_fc1 = (const float*)d_in[10];
  const float* w_fc2 = (const float*)d_in[11];
  const float* b_fc2 = (const float*)d_in[12];
  float* out = (float*)d_out;

  // ---- workspace layout (~118 MB, with region reuse) ----
  char* ws = (char*)d_ws;
  size_t off = 0;
  auto alloc = [&](size_t bytes) {
    size_t o = off; off += (bytes + 255) & ~(size_t)255; return o;
  };
  bf16_t* wqkvT = (bf16_t*)(ws + alloc((size_t)3 * DMODEL * DMODEL * 2));
  bf16_t* woutT = (bf16_t*)(ws + alloc((size_t)DMODEL * DMODEL * 2));
  bf16_t* wfc1T = (bf16_t*)(ws + alloc((size_t)DMLP * DMODEL * 2));
  bf16_t* wfc2T = (bf16_t*)(ws + alloc((size_t)DMODEL * DMLP * 2));
  float*  x1    = (float*)(ws + alloc((size_t)TTOK * DMODEL * 4));
  bf16_t* hA    = (bf16_t*)(ws + alloc((size_t)TTOKP * DMODEL * 2)); // LN out (row-padded)
  size_t qBytes  = (size_t)16 * NHEAD * NPADQ * DHEAD * 2;
  size_t kBytes  = (size_t)16 * NHEAD * NPAD  * DHEAD * 2;
  size_t waBytes = (size_t)TTOKP * DMODEL * 2;    // row-padded (A of out-proj)
  size_t hmBytes = (size_t)TTOKP * DMLP * 2;      // row-padded (A of FC2)
  size_t bsum = qBytes + 2 * kBytes + waBytes;
  char* regB = ws + alloc(bsum > hmBytes ? bsum : hmBytes);
  bf16_t* Qb = (bf16_t*)(regB);
  bf16_t* Kb = (bf16_t*)(regB + qBytes);
  bf16_t* Vt = (bf16_t*)(regB + qBytes + kBytes);
  bf16_t* wa = (bf16_t*)(regB + qBytes + 2 * kBytes);
  bf16_t* hm = (bf16_t*)(regB);        // reuses Q/K/V/wa region after attention+proj

  // ---- weight convert + transpose (bf16) ----
  auto cvt = [&](const float* w, bf16_t* wt, int K, int N) {
    size_t total = (size_t)K * N;
    int blocks = (int)((total + 255) / 256);
    cvt_t_kernel<<<blocks, 256, 0, stream>>>(w, wt, K, N);
  };
  cvt(w_qkv, wqkvT, DMODEL, 3 * DMODEL);
  cvt(w_out, woutT, DMODEL, DMODEL);
  cvt(w_fc1, wfc1T, DMODEL, DMLP);
  cvt(w_fc2, wfc2T, DMLP, DMODEL);

  const int gx = (TTOK + 255) / 256;   // 37 row-blocks of 256 rows

  // ---- LN1 -> hA ----
  ln_kernel<<<TTOK, 256, 0, stream>>>(x, ln1_g, ln1_b, hA);

  // ---- QKV GEMM (M=9232, N=2304, K=768) + scatter to Q/K/Vt ----
  gemm_kernel<EPI_QKV><<<dim3(gx, 36), 256, 0, stream>>>(
      hA, wqkvT, b_qkv, TTOK, 3 * DMODEL, DMODEL,
      nullptr, nullptr, nullptr, Qb, Kb, Vt);

  // ---- fused flash attention -> wa ----
  attn_kernel<<<dim3(16 * NHEAD, NPADQ / 64), 128, 0, stream>>>(Qb, Kb, Vt, wa);

  // ---- out-proj + residual: x1 = x + wa @ w_out + b_out ----
  gemm_kernel<EPI_RESID><<<dim3(gx, 12), 256, 0, stream>>>(
      wa, woutT, b_out, TTOK, DMODEL, DMODEL,
      x, x1, nullptr, nullptr, nullptr, nullptr);

  // ---- LN2 -> hA ----
  ln_kernel<<<TTOK, 256, 0, stream>>>(x1, ln2_g, ln2_b, hA);

  // ---- FC1 + exact GELU -> hm ----
  gemm_kernel<EPI_GELU><<<dim3(gx, 48), 256, 0, stream>>>(
      hA, wfc1T, b_fc1, TTOK, DMLP, DMODEL,
      nullptr, nullptr, hm, nullptr, nullptr, nullptr);

  // ---- FC2 + residual -> out ----
  gemm_kernel<EPI_RESID><<<dim3(gx, 12), 256, 0, stream>>>(
      hm, wfc2T, b_fc2, TTOK, DMODEL, DMLP,
      x1, out, nullptr, nullptr, nullptr, nullptr);
}
